// SharedMamba_42958262894625
// MI455X (gfx1250) — matrix operand order repaired and checked by reading, not compiled, not run on hardware
//
#include <hip/hip_runtime.h>
#include <math.h>

// ---------------- problem constants (match reference) ----------------
#define D_MODEL  1024
#define D_STATE  16
#define D_INNER  2048
#define D_CONV   4
#define DT_RANK  64
#define B_SZ     2
#define L_SEQ    2048
#define NROWS    (B_SZ * L_SEQ)          // 4096 flattened (b,l) rows
#define N_LAYERS 4
#define XZ_W     (2 * D_INNER)           // 4096
#define DBC_W    (DT_RANK + 2 * D_STATE) // 96

typedef __attribute__((ext_vector_type(16))) __bf16 v16bf;
typedef __attribute__((ext_vector_type(8)))  float  v8f;
typedef __attribute__((ext_vector_type(2)))  float  f32x2;
typedef __attribute__((ext_vector_type(2)))  __bf16 bf16x2;

// packed fp32x2 -> bf16x2 (RNE) -- should map to v_cvt_pk_bf16_f32
__device__ __forceinline__ unsigned int pkbf2(float a, float b) {
  f32x2 f = {a, b};
  bf16x2 v = __builtin_convertvector(f, bf16x2);
  return __builtin_bit_cast(unsigned int, v);
}
// scalar fp32 -> bf16 (RNE)
__device__ __forceinline__ unsigned short f2bf(float f) {
  __bf16 h = (__bf16)f;
  return __builtin_bit_cast(unsigned short, h);
}
// bf16 -> fp32 (exact)
__device__ __forceinline__ float bff(unsigned short u) {
  return __uint_as_float(((unsigned int)u) << 16);
}
__device__ __forceinline__ float sigmoidf_(float x) { return 1.f / (1.f + __expf(-x)); }

enum { EPI_NONE = 0, EPI_BIAS_SOFTPLUS = 1, EPI_RESIDUAL = 2 };
enum { OUT_F32 = 0, OUT_BF16 = 1, OUT_BOTH = 2 };

union BFfrag { v16bf v; uint4 q[2]; };

// =====================================================================
// packed fp32 -> bf16 converter (weights + layer-0 input), 2 elems/thread
// =====================================================================
__global__ __launch_bounds__(256)
void cvt_bf16_kernel(const float* __restrict__ src, unsigned int* __restrict__ dst, long npairs)
{
  const long i = (long)blockIdx.x * 256 + threadIdx.x;
  if (i < npairs) {
    const float2 f = ((const float2*)src)[i];
    dst[i] = pkbf2(f.x, f.y);
  }
}

// =====================================================================
// bf16 WMMA GEMM:  C[M,N] = A[M,K] * B[N,K]^T  (+ epilogue)
// A,B bf16 in global; staging to LDS is a pure copy (no VALU conversion).
// C written as fp32 (Cf) and/or bf16 mirror (Ch) per OUT.
// Block: 256 threads, 128x128 C tile, 8 waves (2x4), 64x32 per wave.
// Requires K % 32 == 0 (true here: 1024, 2048, 64).
// =====================================================================
template<int EPI, int OUT>
__global__ __launch_bounds__(256)
void wmma_gemm_kernel(const unsigned short* __restrict__ A, int lda,
                      const unsigned short* __restrict__ Bw, int ldb,
                      float* __restrict__ Cf, unsigned short* __restrict__ Ch, int ldc,
                      int M, int N, int K,
                      const float* __restrict__ aux, int ldaux)
{
  constexpr int TM = 128, TN = 128, TK = 32, LDST = 40; // 40-half row pad: conflict-free frag reads
  __shared__ unsigned short sA[TM * LDST];
  __shared__ unsigned short sB[TN * LDST];

  const int tid   = threadIdx.x;
  const int lane  = tid & 31;
  const int wave  = tid >> 5;
  const int waveM = wave >> 2;     // 0..1
  const int waveN = wave & 3;      // 0..3
  const int row16 = lane & 15;
  const int hi    = lane >> 4;     // 0/1
  const int kbA   = hi * 8;        // A frag: K chunks {kbA..kbA+7, kbA+16..kbA+23}
  const int kbB   = hi * 8;        // B frag: K chunks {kbB..kbB+7, kbB+16..kbB+23}

  const int mBase = blockIdx.y * TM;
  const int nBase = blockIdx.x * TN;

  v8f acc[4][2];
  #pragma unroll
  for (int f = 0; f < 4; ++f)
    #pragma unroll
    for (int g = 0; g < 2; ++g)
      #pragma unroll
      for (int i = 0; i < 8; ++i) acc[f][g][i] = 0.f;

  const int sRow  = tid >> 1;  // 0..127
  const int sHalf = tid & 1;   // 16-col (bf16) half = 32 bytes

  const int gra = mBase + sRow;
  const int grb = nBase + sRow;
  const unsigned short* srcA = A  + (size_t)gra * lda + sHalf * 16;
  const unsigned short* srcB = Bw + (size_t)grb * ldb + sHalf * 16;
  uint4* dstA = (uint4*)(sA + sRow * LDST + sHalf * 16);
  uint4* dstB = (uint4*)(sB + sRow * LDST + sHalf * 16);

  for (int k0 = 0; k0 < K; k0 += TK) {
    __syncthreads();
    if (gra < M) {
      const uint4* s = (const uint4*)(srcA + k0);
      dstA[0] = s[0]; dstA[1] = s[1];
    } else { dstA[0] = make_uint4(0,0,0,0); dstA[1] = make_uint4(0,0,0,0); }
    if (grb < N) {
      const uint4* s = (const uint4*)(srcB + k0);
      dstB[0] = s[0]; dstB[1] = s[1];
    } else { dstB[0] = make_uint4(0,0,0,0); dstB[1] = make_uint4(0,0,0,0); }
    // prefetch next k-tile while this one is consumed (global_prefetch_b8)
    if (k0 + TK < K) {
      if (gra < M) __builtin_prefetch(srcA + k0 + TK, 0, 1);
      if (grb < N) __builtin_prefetch(srcB + k0 + TK, 0, 1);
    }
    __syncthreads();

    BFfrag fa[4], fb[2];
    #pragma unroll
    for (int f = 0; f < 4; ++f) {
      const int r = (waveM * 64 + f * 16 + row16) * LDST;
      fa[f].q[0] = *(const uint4*)(sA + r + kbA);
      fa[f].q[1] = *(const uint4*)(sA + r + kbA + 16);
    }
    #pragma unroll
    for (int g = 0; g < 2; ++g) {
      const int c = (waveN * 32 + g * 16 + row16) * LDST;
      fb[g].q[0] = *(const uint4*)(sB + c + kbB);
      fb[g].q[1] = *(const uint4*)(sB + c + kbB + 16);
    }
    #pragma unroll
    for (int f = 0; f < 4; ++f)
      #pragma unroll
      for (int g = 0; g < 2; ++g)
        acc[f][g] = __builtin_amdgcn_wmma_f32_16x16x32_bf16(
            false, fa[f].v, false, fb[g].v, (short)0, acc[f][g], false, false);
  }

  // epilogue + store. C/D layout: VGPR v -> M = v (+8 for hi lanes), N = lane%16
  #pragma unroll
  for (int g = 0; g < 2; ++g) {
    const int n = nBase + waveN * 32 + g * 16 + row16;
    if (n >= N) continue;
    #pragma unroll
    for (int f = 0; f < 4; ++f) {
      const int m0 = mBase + waveM * 64 + f * 16 + hi * 8;
      #pragma unroll
      for (int v = 0; v < 8; ++v) {
        const int m = m0 + v;
        if (m < M) {
          float val = acc[f][g][v];
          if constexpr (EPI == EPI_BIAS_SOFTPLUS) {
            val += aux[n];
            val = (val > 20.f) ? val : log1pf(__expf(val));
          } else if constexpr (EPI == EPI_RESIDUAL) {
            val += aux[(size_t)m * ldaux + n];
          }
          if constexpr (OUT == OUT_F32 || OUT == OUT_BOTH)
            Cf[(size_t)m * ldc + n] = val;
          if constexpr (OUT == OUT_BF16 || OUT == OUT_BOTH)
            Ch[(size_t)m * ldc + n] = f2bf(val);
        }
      }
    }
  }
}

// =====================================================================
// depthwise causal conv (4 taps) + bias + SiLU, reading xr = xz[...,:2048]
// (xz stored bf16); writes xc as fp32 (for scan) + bf16 (for GEMMs)
// =====================================================================
__global__ __launch_bounds__(256)
void conv_silu_kernel(const unsigned short* __restrict__ xzbf,
                      const float* __restrict__ cw, const float* __restrict__ cb,
                      float* __restrict__ xc, unsigned short* __restrict__ xcbf)
{
  const long idx = (long)blockIdx.x * blockDim.x + threadIdx.x;
  if (idx >= (long)NROWS * D_INNER) return;
  const int  d   = (int)(idx % D_INNER);
  const long row = idx / D_INNER;            // b*L + l
  const int  l   = (int)(row % L_SEQ);
  float acc = cb[d];
  #pragma unroll
  for (int j = 0; j < D_CONV; ++j) {
    const int ls = l - (D_CONV - 1) + j;
    if (ls >= 0)
      acc = fmaf(cw[d * D_CONV + j],
                 bff(xzbf[(row + (long)(ls - l)) * (long)XZ_W + d]), acc);
  }
  const float s = acc * sigmoidf_(acc);
  xc[idx]   = s;
  xcbf[idx] = f2bf(s);
}

// =====================================================================
// selective scan, fused with  y = (scan + D*u) * silu(z)
// One lane per (d,n): scalar recurrence + 16-lane shfl_xor reduction.
// 256 threads = 8 waves = 16 d-channels per block; B/C/dt/u LDS-staged.
// Output y written bf16 (feeds the W_out GEMM's A directly).
// =====================================================================
#define SC_D 16
#define SC_T 128
__global__ __launch_bounds__(256)
void scan_kernel(const float* __restrict__ dbc, const float* __restrict__ dt,
                 const float* __restrict__ xc, const unsigned short* __restrict__ xzbf,
                 const float* __restrict__ A_log, const float* __restrict__ D_skip,
                 unsigned short* __restrict__ ybf)
{
  __shared__ float sBC[SC_T * 32];
  __shared__ float sDT[SC_T * SC_D];
  __shared__ float sU [SC_T * SC_D];
  const int b    = blockIdx.y;
  const int d0   = blockIdx.x * SC_D;
  const int tid  = threadIdx.x;
  const int lane = tid & 31;
  const int wave = tid >> 5;
  const int n    = lane & 15;          // state index
  const int dl   = wave * 2 + (lane >> 4);
  const int d    = d0 + dl;

  const float An  = -__expf(A_log[d * D_STATE + n]);
  const float dsk = D_skip[d];
  float h = 0.f;
  const size_t rowB = (size_t)b * L_SEQ;

  for (int t0 = 0; t0 < L_SEQ; t0 += SC_T) {
    __syncthreads();
    for (int i = tid; i < SC_T * 32; i += 256) {
      const int tt = i >> 5, c = i & 31;
      sBC[i] = dbc[(rowB + t0 + tt) * DBC_W + DT_RANK + c];
    }
    for (int i = tid; i < SC_T * SC_D; i += 256) {
      const int tt = i >> 4, c = i & 15;
      const size_t g = (rowB + t0 + tt) * D_INNER + d0 + c;
      sDT[i] = dt[g];
      sU[i]  = xc[g];
    }
    __syncthreads();
    for (int tt = 0; tt < SC_T; ++tt) {
      const float dtv = sDT[tt * SC_D + dl];
      const float uv  = sU [tt * SC_D + dl];
      const float Bn  = sBC[tt * 32 + n];
      const float Cn  = sBC[tt * 32 + 16 + n];
      h = __expf(dtv * An) * h + (dtv * uv) * Bn;
      float y = h * Cn;
      y += __shfl_xor(y, 1);
      y += __shfl_xor(y, 2);
      y += __shfl_xor(y, 4);
      y += __shfl_xor(y, 8);
      if (n == 0) {
        const size_t row = rowB + t0 + tt;
        const float zv = bff(xzbf[row * XZ_W + D_INNER + d]);
        ybf[row * D_INNER + d] = f2bf((y + dsk * uv) * (zv * sigmoidf_(zv)));
      }
    }
  }
}

// =====================================================================
// final LayerNorm over D_MODEL, one block per row
// =====================================================================
__global__ __launch_bounds__(256)
void layernorm_kernel(const float* __restrict__ x, const float* __restrict__ w,
                      const float* __restrict__ bb, float* __restrict__ out)
{
  __shared__ float rs[256], rq[256];
  const int row = blockIdx.x;
  const float* xr = x + (size_t)row * D_MODEL;
  float s = 0.f, q = 0.f;
  for (int i = threadIdx.x; i < D_MODEL; i += 256) {
    const float v = xr[i]; s += v; q += v * v;
  }
  rs[threadIdx.x] = s; rq[threadIdx.x] = q;
  __syncthreads();
  for (int off = 128; off > 0; off >>= 1) {
    if (threadIdx.x < off) {
      rs[threadIdx.x] += rs[threadIdx.x + off];
      rq[threadIdx.x] += rq[threadIdx.x + off];
    }
    __syncthreads();
  }
  const float mu  = rs[0] * (1.f / D_MODEL);
  const float var = rq[0] * (1.f / D_MODEL) - mu * mu;
  const float inv = rsqrtf(var + 1e-5f);
  float* o = out + (size_t)row * D_MODEL;
  for (int i = threadIdx.x; i < D_MODEL; i += 256)
    o[i] = (xr[i] - mu) * inv * w[i] + bb[i];
}

// =====================================================================
extern "C" void kernel_launch(void* const* d_in, const int* in_sizes, int n_in,
                              void* d_out, int out_size, void* d_ws, size_t ws_size,
                              hipStream_t stream)
{
  (void)in_sizes; (void)n_in; (void)out_size; (void)ws_size;
  const float* x      = (const float*)d_in[0];
  const float* W_in   = (const float*)d_in[1];
  const float* conv_w = (const float*)d_in[2];
  const float* conv_b = (const float*)d_in[3];
  const float* W_x    = (const float*)d_in[4];
  const float* W_dt   = (const float*)d_in[5];
  const float* b_dt   = (const float*)d_in[6];
  const float* A_log  = (const float*)d_in[7];
  const float* D_skip = (const float*)d_in[8];
  const float* W_out  = (const float*)d_in[9];
  const float* ln_w   = (const float*)d_in[10];
  const float* ln_b   = (const float*)d_in[11];

  // ---- workspace layout: fp32 region (~102 MB) then bf16 region (~90 MB) ----
  float* fw = (float*)d_ws;
  float* xc  = fw; fw += (size_t)NROWS * D_INNER;   // 32 MB
  float* dbc = fw; fw += (size_t)NROWS * DBC_W;     // 1.5 MB
  float* dt  = fw; fw += (size_t)NROWS * D_INNER;   // 32 MB
  float* X1  = fw; fw += (size_t)NROWS * D_MODEL;   // 16 MB
  float* X2  = fw; fw += (size_t)NROWS * D_MODEL;   // 16 MB
  unsigned short* hw = (unsigned short*)fw;
  unsigned short* xzbf   = hw; hw += (size_t)NROWS * XZ_W;      // 33.5 MB [xr | z]
  unsigned short* xbf    = hw; hw += (size_t)NROWS * D_MODEL;   // 8 MB (layer input, bf16)
  unsigned short* xcbf   = hw; hw += (size_t)NROWS * D_INNER;   // 16 MB
  unsigned short* dbcbf  = hw; hw += (size_t)NROWS * DBC_W;     // 0.75 MB
  unsigned short* ybf    = hw; hw += (size_t)NROWS * D_INNER;   // 16 MB
  unsigned short* Winbf  = hw; hw += (size_t)XZ_W * D_MODEL;    // 8 MB
  unsigned short* Wxbf   = hw; hw += (size_t)DBC_W * D_INNER;   // 0.38 MB
  unsigned short* Wdtbf  = hw; hw += (size_t)D_INNER * DT_RANK; // 0.25 MB
  unsigned short* Woutbf = hw;                                  // 4 MB

  const dim3 blk(256);
  // one-time (per launch) packed bf16 conversions: weights + layer-0 input
  {
    const long nx  = (long)NROWS * D_MODEL / 2;
    const long nwi = (long)XZ_W * D_MODEL / 2;
    const long nwx = (long)DBC_W * D_INNER / 2;
    const long nwd = (long)D_INNER * DT_RANK / 2;
    const long nwo = (long)D_MODEL * D_INNER / 2;
    cvt_bf16_kernel<<<dim3((unsigned)(nx  / 256)), blk, 0, stream>>>(x,     (unsigned int*)xbf,    nx);
    cvt_bf16_kernel<<<dim3((unsigned)(nwi / 256)), blk, 0, stream>>>(W_in,  (unsigned int*)Winbf,  nwi);
    cvt_bf16_kernel<<<dim3((unsigned)(nwx / 256)), blk, 0, stream>>>(W_x,   (unsigned int*)Wxbf,   nwx);
    cvt_bf16_kernel<<<dim3((unsigned)(nwd / 256)), blk, 0, stream>>>(W_dt,  (unsigned int*)Wdtbf,  nwd);
    cvt_bf16_kernel<<<dim3((unsigned)(nwo / 256)), blk, 0, stream>>>(W_out, (unsigned int*)Woutbf, nwo);
  }

  const float* cur = x;  // fp32 residual stream; xbf is its bf16 mirror
  for (int layer = 0; layer < N_LAYERS; ++layer) {
    float* nxt = (layer & 1) ? X2 : X1;
    // xz = cur @ W_in^T   (4096 x 4096 x 1024), bf16 output only
    wmma_gemm_kernel<EPI_NONE, OUT_BF16><<<dim3(XZ_W / 128, NROWS / 128), blk, 0, stream>>>(
        xbf, D_MODEL, Winbf, D_MODEL, nullptr, xzbf, XZ_W, NROWS, XZ_W, D_MODEL, nullptr, 0);
    // xc = silu(depthwise_conv(xr) + conv_b), fp32 + bf16
    conv_silu_kernel<<<dim3((unsigned)(((long)NROWS * D_INNER) / 256)), blk, 0, stream>>>(
        xzbf, conv_w, conv_b, xc, xcbf);
    // dbc = xc @ W_x^T    (4096 x 96 x 2048), fp32 (scan) + bf16 (dt GEMM)
    wmma_gemm_kernel<EPI_NONE, OUT_BOTH><<<dim3((DBC_W + 127) / 128, NROWS / 128), blk, 0, stream>>>(
        xcbf, D_INNER, Wxbf, D_INNER, dbc, dbcbf, DBC_W, NROWS, DBC_W, D_INNER, nullptr, 0);
    // dt = softplus(dt_low @ W_dt^T + b_dt)  (4096 x 2048 x 64; lda=96 slices dbc[:,:64])
    wmma_gemm_kernel<EPI_BIAS_SOFTPLUS, OUT_F32><<<dim3(D_INNER / 128, NROWS / 128), blk, 0, stream>>>(
        dbcbf, DBC_W, Wdtbf, DT_RANK, dt, nullptr, D_INNER, NROWS, D_INNER, DT_RANK, b_dt, 0);
    // selective scan + D-skip + silu(z) gate -> y (bf16)
    scan_kernel<<<dim3(D_INNER / SC_D, B_SZ), blk, 0, stream>>>(
        dbc, dt, xc, xzbf, A_log, D_skip, ybf);
    // nxt = y @ W_out^T + cur   (4096 x 1024 x 2048); fp32 stream + bf16 mirror for next layer
    wmma_gemm_kernel<EPI_RESIDUAL, OUT_BOTH><<<dim3(D_MODEL / 128, NROWS / 128), blk, 0, stream>>>(
        ybf, D_INNER, Woutbf, D_INNER, nxt, xbf, D_MODEL, NROWS, D_MODEL, D_INNER, cur, D_MODEL);
    cur = nxt;
  }
  layernorm_kernel<<<dim3(NROWS), blk, 0, stream>>>(cur, ln_w, ln_b, (float*)d_out);
}
